// Light_VSSBlock_68023692034382
// MI455X (gfx1250) — matrix operand order repaired and checked by reading, not compiled, not run on hardware
//
#include <hip/hip_runtime.h>

// ---------------- problem constants ----------------
constexpr int BATCH = 16;
constexpr int CHN   = 128;
constexpr int LSEQ  = 64 * 64;     // 4096
constexpr int DM    = 32;          // d_model (chunk width)
constexpr int DSN   = 16;          // d_state
constexpr int DI    = 64;          // d_inner
constexpr int NG    = 4 * BATCH;   // 64 sequences
constexpr float EPSV = 1e-5f;

// ---------------- WMMA plumbing (gfx1250, wave32) ----------------
typedef __attribute__((ext_vector_type(16))) __bf16 v16bf;
typedef __attribute__((ext_vector_type(8)))  float  v8f;

__device__ inline __bf16 to_bf16(float f) {
  unsigned u = __builtin_bit_cast(unsigned, f);
  unsigned r = u + 0x7FFFu + ((u >> 16) & 1u);   // round-to-nearest-even
  return __builtin_bit_cast(__bf16, (unsigned short)(r >> 16));
}

__device__ inline v8f wmma_bf16(v16bf a, v16bf b, v8f c) {
  // D(16x16,f32) = A(16x32,bf16) * B(32x16,bf16) + C
  return __builtin_amdgcn_wmma_f32_16x16x32_bf16(
      /*neg_a=*/false, a, /*neg_b=*/false, b,
      /*c_mod=*/(short)0, c, /*reuse_a=*/false, /*reuse_b=*/false);
}

// A-fragment element e (0..15) of lane:  K = kb + e,                 kb = (lane<16)?0:16, M = lane&15
// B-fragment element e (0..15) of lane:  K = kb + e,                 kb = (lane<16)?0:16, N = lane&15
// D element r (0..7) of lane:            M = r + (lane<16?0:8),      N = lane&15

// =====================================================================
// K1: LayerNorm(C=128) + chunk + in_proj GEMM (WMMA) -> xi,(G,L,64) z,(G,L,64), xn chunks (G,L,32)
// block = 128 threads (4 waves), tile = 64 tokens of one batch b
// =====================================================================
__global__ __launch_bounds__(128) void k1_ln_inproj(
    const float* __restrict__ x,        // (B,C,L)
    const float* __restrict__ ln_g, const float* __restrict__ ln_b,
    const float* __restrict__ in_proj_w,// (128,32)
    float* __restrict__ xi, float* __restrict__ z, float* __restrict__ xnc)
{
  __shared__ float xs[CHN][65];           // +1 pad: conflict-free column walks
  __shared__ float mu_s[64], rs_s[64];
  __shared__ float wlds[CHN * DM];        // 16 KB

  const int b  = blockIdx.y;
  const int l0 = blockIdx.x * 64;
  const int tid = threadIdx.x;

  for (int idx = tid; idx < CHN * 64; idx += 128) {
    int ch = idx >> 6, l = idx & 63;
    xs[ch][l] = x[(size_t)b * CHN * LSEQ + (size_t)ch * LSEQ + l0 + l];
  }
  for (int idx = tid; idx < CHN * DM; idx += 128) wlds[idx] = in_proj_w[idx];
  __syncthreads();

  if (tid < 64) {
    float s = 0.f, s2 = 0.f;
    for (int ch = 0; ch < CHN; ++ch) { float v = xs[ch][tid]; s += v; s2 += v * v; }
    float m = s * (1.f / CHN);
    float var = s2 * (1.f / CHN) - m * m;
    mu_s[tid] = m;
    rs_s[tid] = rsqrtf(var + EPSV);
  }
  __syncthreads();
  for (int idx = tid; idx < CHN * 64; idx += 128) {
    int ch = idx >> 6, l = idx & 63;
    xs[ch][l] = (xs[ch][l] - mu_s[l]) * rs_s[l] * ln_g[ch] + ln_b[ch];
  }
  __syncthreads();

  // store normalized chunks (residual source), (G,L,32), g = c*BATCH + b
  for (int idx = tid; idx < 4 * 64 * DM; idx += 128) {
    int d = idx & 31, l = (idx >> 5) & 63, c = idx >> 11;
    xnc[((size_t)(c * BATCH + b) * LSEQ + l0 + l) * DM + d] = xs[c * DM + d][l];
  }

  // GEMM: wave w -> token rows w*16..w*16+15; loop chunks c (4) x N-tiles (8)
  const int wv = tid >> 5, lane = tid & 31;
  const int lm = lane & 15;
  const int akb = (lane < 16) ? 0 : 16;
  const int bkb = (lane < 16) ? 0 : 16;
  const int mrow = wv * 16 + lm;

  for (int c = 0; c < 4; ++c) {
    v16bf afrag;
#pragma unroll
    for (int e = 0; e < 16; ++e) {
      int k = akb + e;
      afrag[e] = to_bf16(xs[c * DM + k][mrow]);
    }
    const int g = c * BATCH + b;
    for (int nt = 0; nt < 8; ++nt) {
      v16bf bfrag;
#pragma unroll
      for (int e = 0; e < 16; ++e)
        bfrag[e] = to_bf16(wlds[(nt * 16 + lm) * DM + bkb + e]);
      v8f acc = {};
      acc = wmma_bf16(afrag, bfrag, acc);
      const int nglob = nt * 16 + lm;            // output feature (0..127)
      float* dst = (nglob < DI) ? xi : z;
      const int e0 = nglob & (DI - 1);
#pragma unroll
      for (int r = 0; r < 8; ++r) {
        int m = wv * 16 + r + ((lane < 16) ? 0 : 8);
        dst[((size_t)g * LSEQ + l0 + m) * DI + e0] = acc[r];
      }
    }
  }
}

// =====================================================================
// K2: causal depthwise conv1d(k=4) + bias + SiLU -> u (G,L,64)
// =====================================================================
__global__ __launch_bounds__(256) void k2_conv_silu(
    const float* __restrict__ xi, const float* __restrict__ conv_w,
    const float* __restrict__ conv_b, float* __restrict__ u)
{
  size_t idx = (size_t)blockIdx.x * 256 + threadIdx.x;   // over G*L*DI
  int e = (int)(idx & (DI - 1));
  size_t gl = idx >> 6;
  int l = (int)(gl & (LSEQ - 1));
  float acc = conv_b[e];
#pragma unroll
  for (int j = 0; j < 4; ++j) {
    int ls = l - 3 + j;
    if (ls >= 0) acc += conv_w[e * 4 + j] * xi[(gl - (size_t)(3 - j)) * DI + e];
  }
  u[idx] = acc / (1.f + __expf(-acc));     // SiLU
}

// =====================================================================
// K3: x_proj GEMM (WMMA, N padded 34->48) + dt_proj + softplus
//     -> delta (G,L,64), Bm (G,L,16), Cm (G,L,16)
// block = 256 threads (8 waves), each wave one 16-row tile of flattened (G*L)
// =====================================================================
__global__ __launch_bounds__(256) void k3_xproj(
    const float* __restrict__ u, const float* __restrict__ x_proj_w,  // (34,64)
    const float* __restrict__ dt_proj_w,  // (64,2)
    const float* __restrict__ dt_proj_b,  // (64)
    float* __restrict__ delta, float* __restrict__ Bm, float* __restrict__ Cm)
{
  __shared__ float wx[48 * DI];       // padded weights, 12 KB
  __shared__ float xd[8][16 * 48];    // per-wave x_dbl tile, 24 KB

  const int tid = threadIdx.x;
  for (int i = tid; i < 48 * DI; i += 256) {
    int n = i >> 6;
    wx[i] = (n < 34) ? x_proj_w[i] : 0.f;
  }
  __syncthreads();

  const int wv = tid >> 5, lane = tid & 31;
  const int lm = lane & 15;
  const int akb = (lane < 16) ? 0 : 16;
  const int bkb = (lane < 16) ? 0 : 16;
  const size_t rowbase = (size_t)blockIdx.x * 128 + (size_t)wv * 16;

  const float* ur = u + (rowbase + lm) * DI;
  v16bf a0, a1;
#pragma unroll
  for (int e = 0; e < 16; ++e) {
    int k = akb + e;
    a0[e] = to_bf16(ur[k]);
    a1[e] = to_bf16(ur[32 + k]);
  }
#pragma unroll
  for (int nt = 0; nt < 3; ++nt) {
    v16bf b0, b1;
#pragma unroll
    for (int e = 0; e < 16; ++e) {
      b0[e] = to_bf16(wx[(nt * 16 + lm) * DI + bkb + e]);
      b1[e] = to_bf16(wx[(nt * 16 + lm) * DI + 32 + bkb + e]);
    }
    v8f acc = {};
    acc = wmma_bf16(a0, b0, acc);
    acc = wmma_bf16(a1, b1, acc);
#pragma unroll
    for (int r = 0; r < 8; ++r) {
      int m = r + ((lane < 16) ? 0 : 8);
      xd[wv][m * 48 + nt * 16 + lm] = acc[r];
    }
  }
  __syncthreads();

  const float* xw = xd[wv];
  for (int i = lane; i < 16 * DSN; i += 32) {
    int m = i >> 4, n = i & 15;
    Bm[(rowbase + m) * DSN + n] = xw[m * 48 + 2 + n];
    Cm[(rowbase + m) * DSN + n] = xw[m * 48 + 18 + n];
  }
  for (int i = lane; i < 16 * DI; i += 32) {
    int m = i >> 6, dch = i & 63;
    float v = xw[m * 48 + 0] * dt_proj_w[dch * 2 + 0]
            + xw[m * 48 + 1] * dt_proj_w[dch * 2 + 1] + dt_proj_b[dch];
    float sp = (v > 20.f) ? v : log1pf(__expf(v));     // softplus
    delta[(rowbase + m) * DI + dch] = sp;
  }
}

// =====================================================================
// K4: selective scan, one block per sequence g. Thread t: d = t>>2, state quad nq = t&3.
// 32-step LDS tiles, register double-buffered pipeline; y-dot reduced via __shfl_xor.
// Writes yz = (y + u*D) * silu(z), (G,L,64)
// =====================================================================
__global__ __launch_bounds__(256) void k4_scan(
    const float* __restrict__ dlt, const float* __restrict__ u,
    const float* __restrict__ Bmat, const float* __restrict__ Cmat,
    const float* __restrict__ zb, const float* __restrict__ A_log,
    const float* __restrict__ Dv, float* __restrict__ yz)
{
  constexpr int TS = 32;
  __shared__ __attribute__((aligned(16))) float s_d[2][TS][DI];
  __shared__ __attribute__((aligned(16))) float s_u[2][TS][DI];
  __shared__ __attribute__((aligned(16))) float s_z[2][TS][DI];
  __shared__ __attribute__((aligned(16))) float s_B[2][TS][DSN];
  __shared__ __attribute__((aligned(16))) float s_C[2][TS][DSN];

  const int g = blockIdx.x, tid = threadIdx.x;
  const int d = tid >> 2, nq = tid & 3;
  const size_t b64 = (size_t)g * LSEQ * DI;
  const size_t b16 = (size_t)g * LSEQ * DSN;

  const float A0 = -__expf(A_log[d * DSN + nq * 4 + 0]);
  const float A1 = -__expf(A_log[d * DSN + nq * 4 + 1]);
  const float A2 = -__expf(A_log[d * DSN + nq * 4 + 2]);
  const float A3 = -__expf(A_log[d * DSN + nq * 4 + 3]);
  const float Dd = Dv[d];
  float h0 = 0.f, h1 = 0.f, h2 = 0.f, h3 = 0.f;

  float rd[8], ru[8], rz[8], rB[2], rC[2];

  auto load_tile = [&](int l0) {
#pragma unroll
    for (int i = 0; i < 8; ++i) {
      int idx = tid + i * 256;                  // 0..2047
      int s = idx >> 6, e = idx & 63;
      size_t off = b64 + (size_t)(l0 + s) * DI + e;
      rd[i] = dlt[off]; ru[i] = u[off]; rz[i] = zb[off];
    }
#pragma unroll
    for (int i = 0; i < 2; ++i) {
      int idx = tid + i * 256;                  // 0..511
      int s = idx >> 4, n = idx & 15;
      size_t off = b16 + (size_t)(l0 + s) * DSN + n;
      rB[i] = Bmat[off]; rC[i] = Cmat[off];
    }
  };
  auto store_tile = [&](int buf) {
#pragma unroll
    for (int i = 0; i < 8; ++i) {
      int idx = tid + i * 256;
      int s = idx >> 6, e = idx & 63;
      s_d[buf][s][e] = rd[i]; s_u[buf][s][e] = ru[i]; s_z[buf][s][e] = rz[i];
    }
#pragma unroll
    for (int i = 0; i < 2; ++i) {
      int idx = tid + i * 256;
      int s = idx >> 4, n = idx & 15;
      s_B[buf][s][n] = rB[i]; s_C[buf][s][n] = rC[i];
    }
  };

  load_tile(0);
  store_tile(0);
  __syncthreads();

  const int NT = LSEQ / TS;     // 128 tiles
  for (int t = 0; t < NT; ++t) {
    const int cur = t & 1;
    if (t + 1 < NT) load_tile((t + 1) * TS);   // global loads fly during compute
    if (t + 2 < NT && tid < TS) {              // touch tile t+2 (gfx1250 global_prefetch_b8)
      size_t pl = (size_t)((t + 2) * TS + tid);
      __builtin_prefetch(dlt  + b64 + pl * DI, 0, 3);
      __builtin_prefetch(u    + b64 + pl * DI, 0, 3);
      __builtin_prefetch(zb   + b64 + pl * DI, 0, 3);
      __builtin_prefetch(Bmat + b16 + pl * DSN, 0, 3);
      __builtin_prefetch(Cmat + b16 + pl * DSN, 0, 3);
    }
#pragma unroll 4
    for (int s = 0; s < TS; ++s) {
      const int l = t * TS + s;
      float dt_ = s_d[cur][s][d];
      float uu  = s_u[cur][s][d];
      float du  = dt_ * uu;
      float4 Bl = *(const float4*)&s_B[cur][s][nq * 4];
      float4 Cl = *(const float4*)&s_C[cur][s][nq * 4];
      h0 = h0 * __expf(dt_ * A0) + du * Bl.x;
      h1 = h1 * __expf(dt_ * A1) + du * Bl.y;
      h2 = h2 * __expf(dt_ * A2) + du * Bl.z;
      h3 = h3 * __expf(dt_ * A3) + du * Bl.w;
      float y = h0 * Cl.x + h1 * Cl.y + h2 * Cl.z + h3 * Cl.w;
      y += __shfl_xor(y, 1);
      y += __shfl_xor(y, 2);
      if (nq == 0) {
        float zz = s_z[cur][s][d];
        float sig = zz / (1.f + __expf(-zz));          // silu(z)
        yz[b64 + (size_t)l * DI + d] = (y + uu * Dd) * sig;
      }
    }
    __syncthreads();
    if (t + 1 < NT) { store_tile((t + 1) & 1); __syncthreads(); }
  }
}

// =====================================================================
// K5: out_proj GEMM (WMMA) + residual + transpose to (B,C,H,W) layout
// block = 256 threads (8 waves), each wave one 16-token tile
// =====================================================================
__global__ __launch_bounds__(256) void k5_outproj(
    const float* __restrict__ yz, const float* __restrict__ out_proj_w, // (32,64)
    const float* __restrict__ xnc, float* __restrict__ out)
{
  __shared__ float wo[DM * DI];        // 8 KB
  __shared__ float tile[8][DM * 16];   // 16 KB (per-wave ch-major transpose buffer)

  const int tid = threadIdx.x;
  for (int i = tid; i < DM * DI; i += 256) wo[i] = out_proj_w[i];
  __syncthreads();

  const int wv = tid >> 5, lane = tid & 31;
  const int lm = lane & 15;
  const int akb = (lane < 16) ? 0 : 16;
  const int bkb = (lane < 16) ? 0 : 16;
  const size_t rowbase = (size_t)blockIdx.x * 128 + (size_t)wv * 16;
  const int g = (int)(rowbase >> 12);            // / LSEQ
  const int l0 = (int)(rowbase & (LSEQ - 1));
  const int c = g >> 4, b = g & 15;

  const float* yr = yz + (rowbase + lm) * DI;
  v16bf a0, a1;
#pragma unroll
  for (int e = 0; e < 16; ++e) {
    int k = akb + e;
    a0[e] = to_bf16(yr[k]);
    a1[e] = to_bf16(yr[32 + k]);
  }
#pragma unroll
  for (int nt = 0; nt < 2; ++nt) {
    v16bf b0, b1;
#pragma unroll
    for (int e = 0; e < 16; ++e) {
      b0[e] = to_bf16(wo[(nt * 16 + lm) * DI + bkb + e]);
      b1[e] = to_bf16(wo[(nt * 16 + lm) * DI + 32 + bkb + e]);
    }
    v8f acc = {};
    acc = wmma_bf16(a0, b0, acc);
    acc = wmma_bf16(a1, b1, acc);
#pragma unroll
    for (int r = 0; r < 8; ++r) {
      int m = r + ((lane < 16) ? 0 : 8);
      int n = nt * 16 + lm;
      tile[wv][n * 16 + m] = acc[r] + xnc[(rowbase + m) * DM + n];  // + residual
    }
  }
  __syncthreads();

  const float* tw = tile[wv];
  for (int i = lane; i < DM * 16; i += 32) {
    int ch = i >> 4, li = i & 15;
    out[((size_t)b * CHN + (size_t)(c * DM + ch)) * LSEQ + l0 + li] = tw[ch * 16 + li];
  }
}

// =====================================================================
extern "C" void kernel_launch(void* const* d_in, const int* in_sizes, int n_in,
                              void* d_out, int out_size, void* d_ws, size_t ws_size,
                              hipStream_t stream)
{
  (void)in_sizes; (void)n_in; (void)out_size; (void)ws_size;
  const float* x         = (const float*)d_in[0];
  const float* ln_g      = (const float*)d_in[1];
  const float* ln_b      = (const float*)d_in[2];
  const float* in_proj_w = (const float*)d_in[3];
  const float* conv_w    = (const float*)d_in[4];
  const float* conv_b    = (const float*)d_in[5];
  const float* x_proj_w  = (const float*)d_in[6];
  const float* dt_proj_w = (const float*)d_in[7];
  const float* dt_proj_b = (const float*)d_in[8];
  const float* A_log     = (const float*)d_in[9];
  const float* Dv        = (const float*)d_in[10];
  const float* out_proj_w= (const float*)d_in[11];
  float* out = (float*)d_out;

  float* ws = (float*)d_ws;
  const size_t GLE = (size_t)NG * LSEQ * DI;    // 16,777,216 floats
  const size_t GLS = (size_t)NG * LSEQ * DSN;
  float* xi    = ws;                  // 64 MiB
  float* z     = xi + GLE;            // 64 MiB
  float* u     = z + GLE;             // 64 MiB
  float* delta = u + GLE;             // 64 MiB
  float* Bm    = delta + GLE;         // 16 MiB
  float* Cm    = Bm + GLS;            // 16 MiB
  float* xnc   = Cm + GLS;            // 32 MiB
  float* yz    = xi;                  // alias: xi is dead after K2

  k1_ln_inproj<<<dim3(LSEQ / 64, BATCH), 128, 0, stream>>>(
      x, ln_g, ln_b, in_proj_w, xi, z, xnc);
  k2_conv_silu<<<(unsigned)(GLE / 256), 256, 0, stream>>>(xi, conv_w, conv_b, u);
  k3_xproj<<<(unsigned)((size_t)NG * LSEQ / 128), 256, 0, stream>>>(
      u, x_proj_w, dt_proj_w, dt_proj_b, delta, Bm, Cm);
  k4_scan<<<NG, 256, 0, stream>>>(delta, u, Bm, Cm, z, A_log, Dv, yz);
  k5_outproj<<<(unsigned)((size_t)NG * LSEQ / 128), 256, 0, stream>>>(
      yz, out_proj_w, xnc, out);
}
